// _RetentionLayer_1683627180233
// MI455X (gfx1250) — matrix operand order repaired and checked by reading, not compiled, not run on hardware
//
#include <hip/hip_runtime.h>

#define D_DIM   1024
#define ROWS_M  (4 * 4096)
#define T_CHUNK 128
#define N_CHUNKS 32
#define STRIPE  32
#define DECAYF  0.95f

typedef __attribute__((ext_vector_type(16))) __bf16 v16bf;
typedef __attribute__((ext_vector_type(8)))  float  v8f;

union FragBF {
  v16bf v;
  uint4 q[2];
  unsigned short u[16];
};

// hardware bf16 conversions (lower to v_cvt_*bf16* on gfx1250)
__device__ __forceinline__ unsigned short f2bf(float f) {
  union { __bf16 b; unsigned short u; } x;
  x.b = (__bf16)f;
  return x.u;
}
__device__ __forceinline__ float bf2f(unsigned short h) {
  union { unsigned short u; __bf16 b; } x;
  x.u = h;
  return (float)x.b;
}

__device__ __forceinline__ v8f wmma_bf(const FragBF& a, const FragBF& b, v8f c) {
  // D(16x16 f32) = A(16x32 bf16) * B(32x16 bf16) + C
  return __builtin_amdgcn_wmma_f32_16x16x32_bf16(
      false, a.v, false, b.v, (short)0, c, false, false);
}

// CDNA5 async global->LDS copy (ASYNCcnt); lds_addr = low 32 bits of generic
// shared pointer == LDS byte offset per the flat-aperture mapping.
__device__ __forceinline__ void async_b128(const unsigned short* lds_dst,
                                           const unsigned short* gsrc) {
  unsigned lds_off = (unsigned)(unsigned long long)lds_dst;
  unsigned long long ga = (unsigned long long)gsrc;
  asm volatile("global_load_async_to_lds_b128 %0, %1, off"
               :: "v"(lds_off), "v"(ga)
               : "memory");
}
__device__ __forceinline__ void wait_async0() {
  asm volatile("s_wait_asynccnt 0" ::: "memory");
}

// ---------------------------------------------------------------------------
// fp32 -> bf16 bulk conversion (8 elements / thread)
// ---------------------------------------------------------------------------
__global__ __launch_bounds__(256) void cvt_f32_bf16(
    const float* __restrict__ src, unsigned short* __restrict__ dst, int n8) {
  const int i = blockIdx.x * 256 + threadIdx.x;
  if (i >= n8) return;
  const float4 f0 = ((const float4*)src)[i * 2];
  const float4 f1 = ((const float4*)src)[i * 2 + 1];
  unsigned short t[8] = {f2bf(f0.x), f2bf(f0.y), f2bf(f0.z), f2bf(f0.w),
                         f2bf(f1.x), f2bf(f1.y), f2bf(f1.z), f2bf(f1.w)};
  ((uint4*)dst)[i] = *(uint4*)t;
}

// ---------------------------------------------------------------------------
// Y[M,1024] = A[M,1024] @ W[1024,1024]^T, both bf16, f32 accumulate.
// EPI: 0 = store bf16; 1 = sigmoid(x + bias) store bf16; 2 = store f32
// 256 threads (8 waves), tile 128x128, K-step 32, double-buffered LDS filled
// by global_load_async_to_lds_b128.
// ---------------------------------------------------------------------------
template <int EPI>
__global__ __launch_bounds__(256) void gemm_nt(
    const unsigned short* __restrict__ A, const unsigned short* __restrict__ W,
    const float* __restrict__ bias, void* __restrict__ outp) {
  __shared__ __align__(16) unsigned short a_lds[2][128 * 32];
  __shared__ __align__(16) unsigned short w_lds[2][128 * 32];

  const int tid  = threadIdx.x;
  const int lane = tid & 31;
  const int wid  = tid >> 5;
  const int wm   = wid >> 1;      // 0..3 : 32-row slab
  const int wn   = wid & 1;       // 0..1 : 64-col slab
  const int hs   = lane >> 4;
  const int l16  = lane & 15;
  const int m0   = blockIdx.x * 128;
  const int n0   = blockIdx.y * 128;

  v8f acc[2][4];
  for (int i = 0; i < 2; ++i)
    for (int j = 0; j < 4; ++j)
      for (int rr = 0; rr < 8; ++rr) acc[i][j][rr] = 0.0f;

  const int sr = tid >> 1;         // staging row 0..127
  const int sc = (tid & 1) * 16;   // staging col 0 / 16

  auto issue = [&](int ks, int buf) {
    const int k0 = ks * 32;
    const unsigned short* ag = A + (size_t)(m0 + sr) * D_DIM + k0 + sc;
    const unsigned short* wg = W + (size_t)(n0 + sr) * D_DIM + k0 + sc;
    async_b128(&a_lds[buf][sr * 32 + sc],     ag);
    async_b128(&a_lds[buf][sr * 32 + sc + 8], ag + 8);
    async_b128(&w_lds[buf][sr * 32 + sc],     wg);
    async_b128(&w_lds[buf][sr * 32 + sc + 8], wg + 8);
  };

  int cur = 0;
  issue(0, 0);
  wait_async0();
  __syncthreads();

  for (int ks = 0; ks < D_DIM / 32; ++ks) {
    if (ks + 1 < D_DIM / 32) issue(ks + 1, cur ^ 1);

    FragBF fbv[4];
#pragma unroll
    for (int nt = 0; nt < 4; ++nt) {
      const int nrow = wn * 64 + nt * 16 + l16;
      fbv[nt].q[0] = *(const uint4*)&w_lds[cur][nrow * 32 + hs * 16];
      fbv[nt].q[1] = *(const uint4*)&w_lds[cur][nrow * 32 + hs * 16 + 8];
    }
#pragma unroll
    for (int mt = 0; mt < 2; ++mt) {
      const int mrow = wm * 32 + mt * 16 + l16;
      FragBF fa;
      fa.q[0] = *(const uint4*)&a_lds[cur][mrow * 32 + hs * 16];
      fa.q[1] = *(const uint4*)&a_lds[cur][mrow * 32 + hs * 16 + 8];
#pragma unroll
      for (int nt = 0; nt < 4; ++nt)
        acc[mt][nt] = wmma_bf(fa, fbv[nt], acc[mt][nt]);
    }

    wait_async0();
    __syncthreads();
    cur ^= 1;
  }

  // ---- epilogue ----
#pragma unroll
  for (int mt = 0; mt < 2; ++mt) {
#pragma unroll
    for (int nt = 0; nt < 4; ++nt) {
      const int gn = n0 + wn * 64 + nt * 16 + l16;
      const float bia = (EPI == 1) ? bias[gn] : 0.0f;
#pragma unroll
      for (int rr = 0; rr < 8; ++rr) {
        const int gm = m0 + wm * 32 + mt * 16 + hs * 8 + rr;
        const float val = acc[mt][nt][rr];
        const size_t off = (size_t)gm * D_DIM + gn;
        if (EPI == 0) {
          ((unsigned short*)outp)[off] = f2bf(val);
        } else if (EPI == 1) {
          const float s = 1.0f / (1.0f + __expf(-(val + bia)));
          ((unsigned short*)outp)[off] = f2bf(s);
        } else {
          ((float*)outp)[off] = val;
        }
      }
    }
  }
}

// ---------------------------------------------------------------------------
// Chunked retention scan. Grid: (32 column stripes, 4 batches).
// Block: 1024 threads = 32 waves. Each wave owns state[d in mg*64..+64)
// x [e in ng*16..+16) of the stripe as 4 f32 WMMA accumulators (registers).
// Dynamic LDS: kT (64KB, reused as bf16 state) + vT (2KB) + ret (16KB).
// ---------------------------------------------------------------------------
__global__ __launch_bounds__(1024) void retention_kernel(
    const unsigned short* __restrict__ qb, const unsigned short* __restrict__ kb,
    const unsigned short* __restrict__ vb, const unsigned short* __restrict__ gb,
    unsigned short* __restrict__ ob) {
  extern __shared__ __align__(16) unsigned short sm[];
  unsigned short* kT = sm;                     // [1024 d][32 t] (reused: state [32 e][1024 d])
  unsigned short* vT = sm + 32768;             // [32 e][32 t]
  float* retbuf = (float*)(sm + 32768 + 1024); // [128 t][32 e]

  const int b    = blockIdx.y;
  const int e0   = blockIdx.x * STRIPE;
  const int tid  = threadIdx.x;
  const int lane = tid & 31;
  const int wid  = tid >> 5;   // 0..31
  const int mg   = wid >> 1;   // 0..15 -> d range [mg*64, +64)
  const int ng   = wid & 1;    // e sub-tile
  const int hs   = lane >> 4;
  const int l16  = lane & 15;

  const size_t baseB = (size_t)b * 4096 * D_DIM;
  const float kscale = (1.0f - DECAYF) / (float)T_CHUNK;  // folded into v

  v8f state[4];
  for (int i = 0; i < 4; ++i)
    for (int rr = 0; rr < 8; ++rr) state[i][rr] = 0.0f;

  for (int c = 0; c < N_CHUNKS; ++c) {
    const int t0 = c * T_CHUNK;
    // state *= DECAY; WMMA then accumulates scaled kv on top
    for (int i = 0; i < 4; ++i)
      for (int rr = 0; rr < 8; ++rr) state[i][rr] *= DECAYF;

    for (int tb = 0; tb < 4; ++tb) {
      const int tBase = t0 + tb * 32;
      // stage k^T : [d][t] (pure transpose copy, no conversion)
      {
        const int tl = tid & 31;
        const int dbase = (tid >> 5) * 32;
        const unsigned short* src = kb + baseB + (size_t)(tBase + tl) * D_DIM + dbase;
        unsigned short vals[32];
        *(uint4*)&vals[0]  = *(const uint4*)(src);
        *(uint4*)&vals[8]  = *(const uint4*)(src + 8);
        *(uint4*)&vals[16] = *(const uint4*)(src + 16);
        *(uint4*)&vals[24] = *(const uint4*)(src + 24);
#pragma unroll
        for (int j = 0; j < 32; ++j) kT[(dbase + j) * 32 + tl] = vals[j];
      }
      // stage v^T stripe : [e][t], scaled by (1-decay)/T (1 element/thread)
      {
        const int el = tid >> 5;
        const int tl = tid & 31;
        const unsigned short raw =
            vb[baseB + (size_t)(tBase + tl) * D_DIM + e0 + el];
        vT[el * 32 + tl] = f2bf(bf2f(raw) * kscale);
      }
      __syncthreads();
      // state[d,e] += sum_t k[t,d] * (scale*v[t,e])
      FragBF fv;
      {
        const int erow = ng * 16 + l16;
        fv.q[0] = *(const uint4*)&vT[erow * 32 + hs * 16];
        fv.q[1] = *(const uint4*)&vT[erow * 32 + hs * 16 + 8];
      }
#pragma unroll
      for (int mt = 0; mt < 4; ++mt) {
        const int drow = mg * 64 + mt * 16 + l16;
        FragBF fk;
        fk.q[0] = *(const uint4*)&kT[drow * 32 + hs * 16];
        fk.q[1] = *(const uint4*)&kT[drow * 32 + hs * 16 + 8];
        state[mt] = wmma_bf(fk, fv, state[mt]);
      }
      __syncthreads();
    }

    // state -> LDS bf16, layout [e_local][d]; packed ds_store_b128
#pragma unroll
    for (int mt = 0; mt < 4; ++mt) {
      const int eloc = ng * 16 + l16;
      const int d0 = mg * 64 + mt * 16 + hs * 8;
      unsigned short t8[8];
#pragma unroll
      for (int rr = 0; rr < 8; ++rr) t8[rr] = f2bf(state[mt][rr]);
      *(uint4*)&kT[eloc * 1024 + d0] = *(uint4*)t8;
    }
    // zero ret accumulation buffer
    for (int i = tid; i < T_CHUNK * STRIPE; i += 1024) retbuf[i] = 0.0f;
    __syncthreads();

    // ret[t,e] partial = q[t, d in mg-range] @ state[d, e in ng-range]
    FragBF sb[2];
#pragma unroll
    for (int ksb = 0; ksb < 2; ++ksb) {
      const int eloc = ng * 16 + l16;
      const int dbase = mg * 64 + ksb * 32 + hs * 16;
      sb[ksb].q[0] = *(const uint4*)&kT[eloc * 1024 + dbase];
      sb[ksb].q[1] = *(const uint4*)&kT[eloc * 1024 + dbase + 8];
    }
#pragma unroll
    for (int tt = 0; tt < 8; ++tt) {
      v8f racc;
      for (int rr = 0; rr < 8; ++rr) racc[rr] = 0.0f;
#pragma unroll
      for (int ksb = 0; ksb < 2; ++ksb) {
        const int trow = t0 + tt * 16 + l16;
        const int db = mg * 64 + ksb * 32 + hs * 16;
        FragBF fq;
        fq.q[0] = *(const uint4*)(qb + baseB + (size_t)trow * D_DIM + db);
        fq.q[1] = *(const uint4*)(qb + baseB + (size_t)trow * D_DIM + db + 8);
        racc = wmma_bf(fq, sb[ksb], racc);
      }
#pragma unroll
      for (int rr = 0; rr < 8; ++rr) {
        const int tl = tt * 16 + hs * 8 + rr;
        const int eloc = ng * 16 + l16;
        atomicAdd(&retbuf[tl * 32 + eloc], racc[rr]);
      }
    }
    __syncthreads();

    // gate + store bf16 (feeds the final projection)
    {
      const int idx = tid * 4;
      const int tl = idx >> 5;
      const int el = idx & 31;
      float rv[4];
      *(float4*)rv = *(const float4*)&retbuf[idx];
      const unsigned short* gsrc = gb + baseB + (size_t)(t0 + tl) * D_DIM + e0 + el;
      unsigned short g4[4];
      *(uint2*)&g4[0] = *(const uint2*)gsrc;
      unsigned short o4[4];
#pragma unroll
      for (int j = 0; j < 4; ++j) o4[j] = f2bf(rv[j] * bf2f(g4[j]));
      *(uint2*)(ob + baseB + (size_t)(t0 + tl) * D_DIM + e0 + el) = *(uint2*)&o4[0];
    }
    __syncthreads();
  }
}

// ---------------------------------------------------------------------------
extern "C" void kernel_launch(void* const* d_in, const int* in_sizes, int n_in,
                              void* d_out, int out_size, void* d_ws, size_t ws_size,
                              hipStream_t stream) {
  (void)in_sizes; (void)n_in; (void)out_size; (void)ws_size;
  const float* h  = (const float*)d_in[0];
  // d_in[1] = chunk_size (int scalar) == 128, baked in
  const float* Wq = (const float*)d_in[2];
  const float* Wk = (const float*)d_in[3];
  const float* Wv = (const float*)d_in[4];
  const float* Wo = (const float*)d_in[5];
  const float* Wg = (const float*)d_in[6];
  const float* bg = (const float*)d_in[7];

  const size_t PLANE = (size_t)ROWS_M * D_DIM;   // 16M elements
  const size_t WPLANE = (size_t)D_DIM * D_DIM;   // 1M elements
  unsigned short* qb  = (unsigned short*)d_ws;
  unsigned short* kb  = qb + PLANE;
  unsigned short* vb  = kb + PLANE;
  unsigned short* gb  = vb + PLANE;
  unsigned short* ob  = gb + PLANE;
  unsigned short* hb  = ob + PLANE;
  unsigned short* wqb = hb + PLANE;
  unsigned short* wkb = wqb + WPLANE;
  unsigned short* wvb = wkb + WPLANE;
  unsigned short* wgb = wvb + WPLANE;
  unsigned short* wob = wgb + WPLANE;

  // bulk fp32 -> bf16 conversions (h + 5 weight matrices)
  cvt_f32_bf16<<<(int)(PLANE / 8 / 256), 256, 0, stream>>>(h, hb, (int)(PLANE / 8));
  cvt_f32_bf16<<<(int)(WPLANE / 8 / 256), 256, 0, stream>>>(Wq, wqb, (int)(WPLANE / 8));
  cvt_f32_bf16<<<(int)(WPLANE / 8 / 256), 256, 0, stream>>>(Wk, wkb, (int)(WPLANE / 8));
  cvt_f32_bf16<<<(int)(WPLANE / 8 / 256), 256, 0, stream>>>(Wv, wvb, (int)(WPLANE / 8));
  cvt_f32_bf16<<<(int)(WPLANE / 8 / 256), 256, 0, stream>>>(Wg, wgb, (int)(WPLANE / 8));
  cvt_f32_bf16<<<(int)(WPLANE / 8 / 256), 256, 0, stream>>>(Wo, wob, (int)(WPLANE / 8));

  dim3 ggrid(ROWS_M / 128, D_DIM / 128);  // 128 x 8

  gemm_nt<0><<<ggrid, 256, 0, stream>>>(hb, wqb, nullptr, qb);
  gemm_nt<0><<<ggrid, 256, 0, stream>>>(hb, wkb, nullptr, kb);
  gemm_nt<0><<<ggrid, 256, 0, stream>>>(hb, wvb, nullptr, vb);
  gemm_nt<1><<<ggrid, 256, 0, stream>>>(hb, wgb, bg, gb);

  const size_t dynLds = (32768 + 1024) * 2 + T_CHUNK * STRIPE * 4;  // 83968 B
  retention_kernel<<<dim3(D_DIM / STRIPE, 4), 1024, dynLds, stream>>>(qb, kb, vb, gb, ob);

  gemm_nt<2><<<ggrid, 256, 0, stream>>>(ob, wob, nullptr, (float*)d_out);
}
